// PatchNCELoss_29411936043016
// MI455X (gfx1250) — matrix operand order repaired and checked by reading, not compiled, not run on hardware
//
#include <hip/hip_runtime.h>

// PatchNCE loss on MI455X (gfx1250, wave32):
//   loss[m] = LSE_n( inv_t * q[m].k[n] ) - inv_t * q[m].k[m],  m,n in [0,8192), D=256
// Flash-style online LSE over S = (1/T) Q K^T using split-bf16 WMMA (hi/lo decomposition).
// M=32 query rows resident per wave (two 16-row A tiles) -> 48 FLOP per loaded B byte.

typedef __attribute__((ext_vector_type(16))) __bf16 v16bf;
typedef __attribute__((ext_vector_type(8)))  __bf16 v8bf;
typedef __attribute__((ext_vector_type(8)))  float  v8f;

#define DK      256                      // feature dim (K of the GEMM)
#define NROWS   8192                     // B*P
#define NTILES  (NROWS / 16)             // 512 key tiles
#define NELEM   (NROWS * DK)             // elements per input tensor
#define INV_T   ((float)(1.0 / 0.07))
#define LN2     0.6931471805599453f
#define SCALE2  ((float)((1.0 / 0.07) * 1.4426950408889634))
#define NEG_INF (-3.0e38f)

// ---------------------------------------------------------------- pass 1 ----
// Split f32 -> bf16 hi + bf16 lo (lo = bf16(x - float(hi))).
__global__ void __launch_bounds__(256)
split_bf16_kernel(const float* __restrict__ x,
                  unsigned short* __restrict__ hi,
                  unsigned short* __restrict__ lo, int n) {
  int i = blockIdx.x * blockDim.x + threadIdx.x;
  if (i < n) {
    float v  = x[i];
    __bf16 h = (__bf16)v;
    __bf16 l = (__bf16)(v - (float)h);
    hi[i] = __builtin_bit_cast(unsigned short, h);
    lo[i] = __builtin_bit_cast(unsigned short, l);
  }
}

// ------------------------------------------------------------- helpers -----
__device__ __forceinline__ v8f wmma_bf16(v16bf a, v16bf b, v8f c) {
  // D = A(16x32 bf16) * B(32x16 bf16) + C(16x16 f32)
  return __builtin_amdgcn_wmma_f32_16x16x32_bf16(false, a, false, b,
                                                 (short)0, c, false, false);
}

// A fragment (ISA 7.12.2, 16-bit A 16x32): lane holds row (lane&15); halves
// 0..7 = K[base..base+7], 8..15 = K[base+16..base+23], base=(lane>>4)*8.
__device__ __forceinline__ v16bf load_a16(const __bf16* p) {
  union { v16bf v; v8bf h[2]; } u;
  u.h[0] = *(const v8bf*)(p);
  u.h[1] = *(const v8bf*)(p + 16);
  return u.v;
}

// B fragment (32x16, column = lane&15): halves 0..7 = K[base..base+7],
// 8..15 = K[base+16..base+23], base=(lane>>4)*8 (same order as A).
__device__ __forceinline__ v16bf load_b16(const __bf16* p) {
  union { v16bf v; v8bf h[2]; } u;
  u.h[0] = *(const v8bf*)(p);
  u.h[1] = *(const v8bf*)(p + 16);
  return u.v;
}

// online LSE update in base-2 domain
__device__ __forceinline__ void lse_update(float x, float& m, float& s) {
  float mn = fmaxf(m, x);
  s = s * __builtin_exp2f(m - mn) + __builtin_exp2f(x - mn);
  m = mn;
}

// ---------------------------------------------------------------- pass 2 ----
// 8 waves / block, 32 query rows per wave (two 16-row tiles).
// 8192/32 = 256 groups -> 32 blocks of 256 threads.
__global__ void __launch_bounds__(256, 1)
patchnce_lse_kernel(const float* __restrict__ qf, const float* __restrict__ kf,
                    const unsigned short* __restrict__ qh_u,
                    const unsigned short* __restrict__ ql_u,
                    const unsigned short* __restrict__ kh_u,
                    const unsigned short* __restrict__ kl_u,
                    float* __restrict__ out) {
  const __bf16* qh = (const __bf16*)qh_u;
  const __bf16* ql = (const __bf16*)ql_u;
  const __bf16* kh = (const __bf16*)kh_u;
  const __bf16* kl = (const __bf16*)kl_u;

  const int lane = threadIdx.x & 31;
  const int wid  = threadIdx.x >> 5;
  const int m0   = (blockIdx.x * 8 + wid) << 5;    // first of 32 query rows

  const int r     = lane & 15;
  const int abase = (lane >> 4) * 8;

  // ---- resident A: 32 rows x 256 K, hi+lo bf16 (2 tiles x 8 chunks) ----
  v16bf Ah0[8], Al0[8], Ah1[8], Al1[8];
  {
    const __bf16* p0h = qh + (m0 + r) * DK + abase;
    const __bf16* p0l = ql + (m0 + r) * DK + abase;
    const __bf16* p1h = qh + (m0 + 16 + r) * DK + abase;
    const __bf16* p1l = ql + (m0 + 16 + r) * DK + abase;
#pragma unroll
    for (int c = 0; c < 8; ++c) {
      Ah0[c] = load_a16(p0h + c * 32);
      Al0[c] = load_a16(p0l + c * 32);
      Ah1[c] = load_a16(p1h + c * 32);
      Al1[c] = load_a16(p1l + c * 32);
    }
  }

  // ---- online LSE state: slot e == local row (e + 8*(lane>>4)) within tile,
  //      column subset {n : n mod 16 == lane&15} ----
  float mr0[8], sr0[8], mr1[8], sr1[8];
#pragma unroll
  for (int e = 0; e < 8; ++e) {
    mr0[e] = NEG_INF; sr0[e] = 0.f;
    mr1[e] = NEG_INF; sr1[e] = 0.f;
  }

  // ---- stream all 512 key tiles, per-block staggered start phase ----
  const int bbase = (lane >> 4) * 8;
  int nt = (blockIdx.x * (NTILES / 32)) & (NTILES - 1);
  const __bf16* bh = kh + (nt * 16 + r) * DK + bbase;
  const __bf16* bl = kl + (nt * 16 + r) * DK + bbase;

  for (int it = 0; it < NTILES; ++it) {
    // prefetch next key tile into near caches (global_prefetch_b8)
    __builtin_prefetch(bh + 16 * DK, 0, 3);
    __builtin_prefetch(bl + 16 * DK, 0, 3);

    v8f a00 = {0.f, 0.f, 0.f, 0.f, 0.f, 0.f, 0.f, 0.f};
    v8f a01 = a00, a02 = a00, a10 = a00, a11 = a00, a12 = a00;
#pragma unroll
    for (int c = 0; c < 8; ++c) {
      v16bf Bh = load_b16(bh + c * 32);
      v16bf Bl = load_b16(bl + c * 32);
      a00 = wmma_bf16(Ah0[c], Bh, a00);   // tile0 hi*hi
      a01 = wmma_bf16(Al0[c], Bh, a01);   // tile0 lo*hi
      a02 = wmma_bf16(Ah0[c], Bl, a02);   // tile0 hi*lo
      a10 = wmma_bf16(Ah1[c], Bh, a10);   // tile1 hi*hi
      a11 = wmma_bf16(Al1[c], Bh, a11);   // tile1 lo*hi
      a12 = wmma_bf16(Ah1[c], Bl, a12);   // tile1 hi*lo
    }

#pragma unroll
    for (int e = 0; e < 8; ++e) {
      lse_update((a00[e] + a01[e] + a02[e]) * SCALE2, mr0[e], sr0[e]);
      lse_update((a10[e] + a11[e] + a12[e]) * SCALE2, mr1[e], sr1[e]);
    }

    bh += 16 * DK;
    bl += 16 * DK;
    if (++nt == NTILES) {               // wrap around the staggered stream
      nt = 0;
      bh -= (size_t)NROWS * DK;
      bl -= (size_t)NROWS * DK;
    }
  }

  // ---- exact f32 diagonals: inv_t * q[row].k[row] ----
  float diag0, diag1;
  {
    const int half = lane >> 4;
    const float4* q0 = (const float4*)(qf + (m0 + r) * DK + half * 128);
    const float4* k0 = (const float4*)(kf + (m0 + r) * DK + half * 128);
    const float4* q1 = (const float4*)(qf + (m0 + 16 + r) * DK + half * 128);
    const float4* k1 = (const float4*)(kf + (m0 + 16 + r) * DK + half * 128);
    float s0 = 0.f, s1 = 0.f;
#pragma unroll 8
    for (int i = 0; i < 32; ++i) {
      float4 a = q0[i], b = k0[i];
      s0 = fmaf(a.x, b.x, fmaf(a.y, b.y, fmaf(a.z, b.z, fmaf(a.w, b.w, s0))));
      float4 c = q1[i], d = k1[i];
      s1 = fmaf(c.x, d.x, fmaf(c.y, d.y, fmaf(c.z, d.z, fmaf(c.w, d.w, s1))));
    }
    s0 += __shfl_xor(s0, 16, 32);
    s1 += __shfl_xor(s1, 16, 32);
    diag0 = s0 * INV_T;
    diag1 = s1 * INV_T;
  }

  // ---- merge the 16 column-partials per row (xor tree within each half) ----
#pragma unroll
  for (int mask = 1; mask <= 8; mask <<= 1) {
#pragma unroll
    for (int e = 0; e < 8; ++e) {
      float mo0 = __shfl_xor(mr0[e], mask, 32);
      float so0 = __shfl_xor(sr0[e], mask, 32);
      float mn0 = fmaxf(mr0[e], mo0);
      sr0[e] = sr0[e] * __builtin_exp2f(mr0[e] - mn0) +
               so0    * __builtin_exp2f(mo0 - mn0);
      mr0[e] = mn0;
      float mo1 = __shfl_xor(mr1[e], mask, 32);
      float so1 = __shfl_xor(sr1[e], mask, 32);
      float mn1 = fmaxf(mr1[e], mo1);
      sr1[e] = sr1[e] * __builtin_exp2f(mr1[e] - mn1) +
               so1    * __builtin_exp2f(mo1 - mn1);
      mr1[e] = mn1;
    }
  }

  // tile0: lane0 holds rows m0+0..7 (slot e -> row e), lane16 rows m0+8..15;
  // tile1: same +16.
#pragma unroll
  for (int e = 0; e < 8; ++e) {
    float lse0 = LN2 * (mr0[e] + __log2f(sr0[e]));
    float lse1 = LN2 * (mr1[e] + __log2f(sr1[e]));
    float d0lo = __shfl(diag0, e, 32);
    float d0hi = __shfl(diag0, 8 + e, 32);
    float d1lo = __shfl(diag1, e, 32);
    float d1hi = __shfl(diag1, 8 + e, 32);
    if (lane == 0) {
      out[m0 + e]      = lse0 - d0lo;
      out[m0 + 16 + e] = lse1 - d1lo;
    }
    if (lane == 16) {
      out[m0 + 8 + e]  = lse0 - d0hi;
      out[m0 + 24 + e] = lse1 - d1hi;
    }
  }
}

// ---------------------------------------------------------------- launch ----
extern "C" void kernel_launch(void* const* d_in, const int* in_sizes, int n_in,
                              void* d_out, int out_size, void* d_ws, size_t ws_size,
                              hipStream_t stream) {
  const float* q = (const float*)d_in[0];   // [32,256,256] f32
  const float* k = (const float*)d_in[1];   // [32,256,256] f32
  float* out = (float*)d_out;               // [8192] f32

  unsigned short* qh = (unsigned short*)d_ws;      // 16 MB total workspace
  unsigned short* ql = qh + NELEM;
  unsigned short* kh = ql + NELEM;
  unsigned short* kl = kh + NELEM;

  const int thr = 256;
  split_bf16_kernel<<<(NELEM + thr - 1) / thr, thr, 0, stream>>>(q, qh, ql, NELEM);
  split_bf16_kernel<<<(NELEM + thr - 1) / thr, thr, 0, stream>>>(k, kh, kl, NELEM);

  // 256 groups of 32 rows; 8 waves per 256-thread block -> 32 blocks
  patchnce_lse_kernel<<<NROWS / 32 / 8, 256, 0, stream>>>(q, k, qh, ql, kh, kl, out);
}
